// Mesh_mamba_seg_70420283785416
// MI455X (gfx1250) — matrix operand order repaired and checked by reading, not compiled, not run on hardware
//
#include <hip/hip_runtime.h>
#include <cstdint>
#include <cstddef>

// ---------------- model constants ----------------
#define BB     8
#define NP     256
#define PP     64
#define CH     13
#define DD     768
#define DEPTH  12
#define DIN    1536
#define NN     16
#define KK     4
#define RR     48
#define SEG    4
#define LSEQ   257                 // NP tokens + cls
#define ROWS_T (BB*LSEQ)           // 2056
#define ROWS_P (BB*NP)             // 2048
#define ROWS_S (BB*NP*PP)          // 131072

// ---------------- WMMA types ----------------
typedef __attribute__((ext_vector_type(16))) __bf16 v16bf;
typedef __attribute__((ext_vector_type(8)))  float  v8f;
union BF16x16 { uint32_t u[8]; v16bf v; };

__device__ __forceinline__ uint16_t f2bf(float f) {
    uint32_t u = __float_as_uint(f);
    u += 0x7FFFu + ((u >> 16) & 1u);   // round-to-nearest-even
    return (uint16_t)(u >> 16);
}
__device__ __forceinline__ uint32_t pk_bf16(float lo, float hi) {
    return (uint32_t)f2bf(lo) | ((uint32_t)f2bf(hi) << 16);
}
__device__ __forceinline__ float gelu_exact(float x) {
    return 0.5f * x * (1.0f + erff(x * 0.70710678118654752f));
}
__device__ __forceinline__ float silu_f(float x) {
    return x / (1.0f + __expf(-x));
}

// =====================================================================
// Generic WMMA GEMM:  C[M,N] = act( A[M,K] @ W[N,K]^T + bias + rowAdd )
//   256 threads = 8 waves; block tile 128(M) x 128(N), K step 32.
//   Wave grid 4(M) x 2(N): each wave computes 32x64 via 2 A-frags x 4 B-frags.
//   Fast path: vectorized float4 loads + packed bf16 ds_store_b64.
//   rowAdd (optional): C[gm,:] += rowAdd[ ((rowBase+gm)>>shA)+((rowBase+gm)>>shB) , :]
//   ACT: 0=none, 1=relu, 4=softplus
// =====================================================================
template <int ACT>
__global__ __launch_bounds__(256)
void gemm_kernel(const float* __restrict__ A, int lda,
                 const float* __restrict__ W, int ldw,
                 const float* __restrict__ bias,
                 const float* __restrict__ rowAdd, int ldadd, int rowBase, int shA, int shB,
                 float* __restrict__ C, int ldc,
                 int M, int N, int Kd)
{
    __shared__ __align__(16) uint16_t lA[128 * 40];
    __shared__ __align__(16) uint16_t lB[128 * 40];

    const int tid  = threadIdx.x;
    const int wave = tid >> 5;
    const int lane = tid & 31;
    const int lr   = lane & 15;
    const int hi   = lane >> 4;
    const int wm   = wave >> 1;        // 0..3 -> M sub-tile base wm*32
    const int wn   = wave & 1;         // 0..1 -> N sub-tile base wn*64
    const int m0   = blockIdx.y * 128;
    const int n0   = blockIdx.x * 128;

    v8f acc[2][4];
#pragma unroll
    for (int f = 0; f < 2; ++f)
#pragma unroll
        for (int t = 0; t < 4; ++t)
#pragma unroll
            for (int e = 0; e < 8; ++e) acc[f][t][e] = 0.0f;

    const bool interiorMN = (m0 + 128 <= M) && (n0 + 128 <= N);

    for (int kk = 0; kk < Kd; kk += 32) {
        if (interiorMN && (kk + 32 <= Kd)) {
            // ---- fast staging: float4 global loads, packed bf16 b64 LDS stores ----
#pragma unroll
            for (int q = 0; q < 4; ++q) {
                int chunk = tid + 256 * q;          // 1024 chunks of 4 elems = 128x32
                int r  = chunk >> 3;
                int kq = (chunk & 7) * 4;
                float4 v = *(const float4*)(A + (size_t)(m0 + r) * lda + kk + kq);
                uint32_t p0 = pk_bf16(v.x, v.y), p1 = pk_bf16(v.z, v.w);
                *(uint2*)&lA[r * 40 + kq] = make_uint2(p0, p1);
            }
#pragma unroll
            for (int q = 0; q < 4; ++q) {
                int chunk = tid + 256 * q;
                int r  = chunk >> 3;
                int kq = (chunk & 7) * 4;
                float4 v = *(const float4*)(W + (size_t)(n0 + r) * ldw + kk + kq);
                uint32_t p0 = pk_bf16(v.x, v.y), p1 = pk_bf16(v.z, v.w);
                *(uint2*)&lB[r * 40 + kq] = make_uint2(p0, p1);
            }
        } else {
            // ---- edge staging: per-element guards ----
            for (int j = tid; j < 128 * 32; j += 256) {
                int r = j >> 5, k = j & 31;
                int gm = m0 + r, gk = kk + k;
                float v = (gm < M && gk < Kd) ? A[(size_t)gm * lda + gk] : 0.0f;
                lA[r * 40 + k] = f2bf(v);
            }
            for (int j = tid; j < 128 * 32; j += 256) {
                int r = j >> 5, k = j & 31;
                int gn = n0 + r, gk = kk + k;
                float v = (gn < N && gk < Kd) ? W[(size_t)gn * ldw + gk] : 0.0f;
                lB[r * 40 + k] = f2bf(v);
            }
        }
        __syncthreads();

        // A fragments (2 per wave): lanes 0-15 -> M=lane, K {2j,2j+1}|{16+2j}; lanes 16-31: K+8
        BF16x16 afr[2];
#pragma unroll
        for (int f = 0; f < 2; ++f) {
            const uint16_t* Ar = &lA[(wm * 32 + f * 16 + lr) * 40];
#pragma unroll
            for (int j = 0; j < 4; ++j)
                afr[f].u[j] = *(const uint32_t*)&Ar[hi * 8 + 2 * j];
#pragma unroll
            for (int j = 0; j < 4; ++j)
                afr[f].u[4 + j] = *(const uint32_t*)&Ar[16 + hi * 8 + 2 * j];
        }

#pragma unroll
        for (int nt = 0; nt < 4; ++nt) {
            // B fragment: lanes 0-15 -> N=lane, K 0..7|16..23; lanes 16-31: K+8
            BF16x16 bfr;
            const uint16_t* Br = &lB[(wn * 64 + nt * 16 + lr) * 40];
#pragma unroll
            for (int j = 0; j < 4; ++j)
                bfr.u[j] = *(const uint32_t*)&Br[hi * 8 + 2 * j];
#pragma unroll
            for (int j = 0; j < 4; ++j)
                bfr.u[4 + j] = *(const uint32_t*)&Br[16 + hi * 8 + 2 * j];
            acc[0][nt] = __builtin_amdgcn_wmma_f32_16x16x32_bf16(
                false, afr[0].v, false, bfr.v, (short)0, acc[0][nt], false, false);
            acc[1][nt] = __builtin_amdgcn_wmma_f32_16x16x32_bf16(
                false, afr[1].v, false, bfr.v, (short)0, acc[1][nt], false, false);
        }
        __syncthreads();
    }

    // epilogue: C/D layout -> VGPR r: lanes0-15 M=r, lanes16-31 M=r+8; N=lane&15
#pragma unroll
    for (int f = 0; f < 2; ++f) {
#pragma unroll
        for (int nt = 0; nt < 4; ++nt) {
            int gn = n0 + wn * 64 + nt * 16 + lr;
#pragma unroll
            for (int r = 0; r < 8; ++r) {
                int gm = m0 + wm * 32 + f * 16 + hi * 8 + r;
                if (gm < M && gn < N) {
                    float v = acc[f][nt][r];
                    if (bias) v += bias[gn];
                    if (rowAdd) {
                        int gr  = rowBase + gm;
                        int idx = (gr >> shA) + (gr >> shB);
                        v += rowAdd[(size_t)idx * ldadd + gn];
                    }
                    if (ACT == 1) v = fmaxf(v, 0.0f);
                    else if (ACT == 4) v = (v > 20.0f) ? v : log1pf(__expf(v));
                    C[(size_t)gm * ldc + gn] = v;
                }
            }
        }
    }
}

// =====================================================================
// LayerNorm over 768 dims, one 256-thread block per row.
// out[orow] = LN(in[row])*g + b (+ addend[row]);  orow = row + (row>>shift) if shift>=0
// =====================================================================
__global__ __launch_bounds__(256)
void ln768_kernel(const float* __restrict__ in, int ldin,
                  const float* __restrict__ g, const float* __restrict__ bt,
                  const float* __restrict__ addend, int ldadd,
                  float* __restrict__ out, int ldout, int shift)
{
    __shared__ float red[256];
    int row = blockIdx.x, t = threadIdx.x;
    float v[3]; float s = 0.0f;
#pragma unroll
    for (int i = 0; i < 3; ++i) { v[i] = in[(size_t)row * ldin + t + 256 * i]; s += v[i]; }
    red[t] = s; __syncthreads();
    for (int o = 128; o > 0; o >>= 1) { if (t < o) red[t] += red[t + o]; __syncthreads(); }
    float mean = red[0] * (1.0f / 768.0f);
    __syncthreads();
    float s2 = 0.0f;
#pragma unroll
    for (int i = 0; i < 3; ++i) { float d = v[i] - mean; s2 += d * d; }
    red[t] = s2; __syncthreads();
    for (int o = 128; o > 0; o >>= 1) { if (t < o) red[t] += red[t + o]; __syncthreads(); }
    float rstd = rsqrtf(red[0] * (1.0f / 768.0f) + 1e-5f);
    int orow = (shift >= 0) ? (row + (row >> shift)) : row;
#pragma unroll
    for (int i = 0; i < 3; ++i) {
        int d = t + 256 * i;
        float o = (v[i] - mean) * rstd * g[d] + bt[d];
        if (addend) o += addend[(size_t)row * ldadd + d];
        out[(size_t)orow * ldout + d] = o;
    }
}

// residual accumulate + LN (per mamba block prologue)
__global__ __launch_bounds__(256)
void residual_ln_kernel(float* __restrict__ residual, const float* __restrict__ tokens,
                        const float* __restrict__ g, const float* __restrict__ bt,
                        float* __restrict__ out, int first)
{
    __shared__ float red[256];
    int row = blockIdx.x, t = threadIdx.x;
    float v[3]; float s = 0.0f;
#pragma unroll
    for (int i = 0; i < 3; ++i) {
        size_t ix = (size_t)row * 768 + t + 256 * i;
        float r = first ? tokens[ix] : (residual[ix] + tokens[ix]);
        residual[ix] = r; v[i] = r; s += r;
    }
    red[t] = s; __syncthreads();
    for (int o = 128; o > 0; o >>= 1) { if (t < o) red[t] += red[t + o]; __syncthreads(); }
    float mean = red[0] * (1.0f / 768.0f);
    __syncthreads();
    float s2 = 0.0f;
#pragma unroll
    for (int i = 0; i < 3; ++i) { float d = v[i] - mean; s2 += d * d; }
    red[t] = s2; __syncthreads();
    for (int o = 128; o > 0; o >>= 1) { if (t < o) red[t] += red[t + o]; __syncthreads(); }
    float rstd = rsqrtf(red[0] * (1.0f / 768.0f) + 1e-5f);
#pragma unroll
    for (int i = 0; i < 3; ++i) {
        int d = t + 256 * i;
        out[(size_t)row * 768 + d] = (v[i] - mean) * rstd * g[d] + bt[d];
    }
}

// face branch stage 1: ts = gelu( LN( feats @ face_w1^T + b1 ) )   (K=13, done in VALU)
__global__ __launch_bounds__(256)
void ts_kernel(const float* __restrict__ feats, const float* __restrict__ w1,
               const float* __restrict__ b1, const float* __restrict__ g,
               const float* __restrict__ bt, float* __restrict__ out)
{
    __shared__ float f13[16];
    __shared__ float red[256];
    int row = blockIdx.x, t = threadIdx.x;
    if (t < CH) f13[t] = feats[(size_t)row * CH + t];
    __syncthreads();
    float v[3]; float s = 0.0f;
#pragma unroll
    for (int i = 0; i < 3; ++i) {
        int d = t + 256 * i;
        float a = b1[d];
        const float* wr = &w1[(size_t)d * CH];
#pragma unroll
        for (int j = 0; j < CH; ++j) a += f13[j] * wr[j];
        v[i] = a; s += a;
    }
    red[t] = s; __syncthreads();
    for (int o = 128; o > 0; o >>= 1) { if (t < o) red[t] += red[t + o]; __syncthreads(); }
    float mean = red[0] * (1.0f / 768.0f);
    __syncthreads();
    float s2 = 0.0f;
#pragma unroll
    for (int i = 0; i < 3; ++i) { float d = v[i] - mean; s2 += d * d; }
    red[t] = s2; __syncthreads();
    for (int o = 128; o > 0; o >>= 1) { if (t < o) red[t] += red[t + o]; __syncthreads(); }
    float rstd = rsqrtf(red[0] * (1.0f / 768.0f) + 1e-5f);
#pragma unroll
    for (int i = 0; i < 3; ++i) {
        int d = t + 256 * i;
        out[(size_t)row * 768 + d] = gelu_exact((v[i] - mean) * rstd * g[d] + bt[d]);
    }
}

// centers mean over P
__global__ void center_kernel(const float* __restrict__ centers, float* __restrict__ center)
{
    int idx = blockIdx.x * blockDim.x + threadIdx.x;
    if (idx >= ROWS_P * 3) return;
    int row = idx / 3, j = idx % 3;
    float s = 0.0f;
    for (int p = 0; p < PP; ++p) s += centers[((size_t)row * PP + p) * 3 + j];
    center[idx] = s * (1.0f / 64.0f);
}

// pos MLP: 3 -> 128 (gelu) -> 768
__global__ __launch_bounds__(128)
void posmlp_kernel(const float* __restrict__ center,
                   const float* __restrict__ w1, const float* __restrict__ b1,
                   const float* __restrict__ w2, const float* __restrict__ b2,
                   float* __restrict__ pos)
{
    __shared__ float hid[128];
    int row = blockIdx.x, t = threadIdx.x;
    float c0 = center[row * 3], c1 = center[row * 3 + 1], c2 = center[row * 3 + 2];
    float hv = b1[t] + c0 * w1[t * 3] + c1 * w1[t * 3 + 1] + c2 * w1[t * 3 + 2];
    hid[t] = gelu_exact(hv);
    __syncthreads();
#pragma unroll
    for (int r = 0; r < 6; ++r) {
        int d = t + 128 * r;
        float s = b2[d];
        const float* wr = &w2[(size_t)d * 128];
        for (int j = 0; j < 128; ++j) s += hid[j] * wr[j];
        pos[(size_t)row * 768 + d] = s;
    }
}

// write cls token rows
__global__ void set_cls_kernel(const float* __restrict__ cls, float* __restrict__ tokens)
{
    int idx = blockIdx.x * blockDim.x + threadIdx.x;
    if (idx >= BB * DD) return;
    int b = idx / DD, d = idx % DD;
    tokens[((size_t)b * LSEQ + NP) * DD + d] = cls[d];
}

// causal depthwise conv (K=4) + bias + SiLU : reads xc = xz[:, :1536]
__global__ void conv_silu_kernel(const float* __restrict__ xz,
                                 const float* __restrict__ w, const float* __restrict__ bias,
                                 float* __restrict__ out)
{
    int idx = blockIdx.x * blockDim.x + threadIdx.x;
    if (idx >= ROWS_T * DIN) return;
    int row = idx / DIN, c = idx % DIN;
    int l = row % LSEQ;
    float acc = bias[c];
    const float* wc = &w[(size_t)c * KK];
#pragma unroll
    for (int k = 0; k < KK; ++k) {
        int ll = l - 3 + k;
        if (ll >= 0) acc += wc[k] * xz[(size_t)(row - 3 + k) * (2 * DIN) + c];
    }
    out[(size_t)row * DIN + c] = silu_f(acc);
}

// selective scan: one thread per (b, channel), sequential over L
__global__ __launch_bounds__(256)
void scan_kernel(const float* __restrict__ dt, const float* __restrict__ xc,
                 const float* __restrict__ xdbl, const float* __restrict__ xz,
                 const float* __restrict__ A_log, const float* __restrict__ Dv,
                 float* __restrict__ ymod)
{
    __shared__ float Bs[NN], Cs[NN];
    int b = blockIdx.y;
    int c = blockIdx.x * 256 + threadIdx.x;
    float An[NN], h[NN];
#pragma unroll
    for (int n = 0; n < NN; ++n) { An[n] = -__expf(A_log[(size_t)c * NN + n]); h[n] = 0.0f; }
    float Dc = Dv[c];
    for (int l = 0; l < LSEQ; ++l) {
        int row = b * LSEQ + l;
        if (threadIdx.x < 32) {
            float v = xdbl[(size_t)row * 80 + RR + threadIdx.x];
            if (threadIdx.x < NN) Bs[threadIdx.x] = v;
            else Cs[threadIdx.x - NN] = v;
        }
        __syncthreads();
        float dtv = dt[(size_t)row * DIN + c];
        float xv  = xc[(size_t)row * DIN + c];
        float acc = 0.0f;
#pragma unroll
        for (int n = 0; n < NN; ++n) {
            float dA = __expf(dtv * An[n]);
            h[n] = dA * h[n] + dtv * Bs[n] * xv;
            acc += h[n] * Cs[n];
        }
        float y  = acc + Dc * xv;
        float zv = xz[(size_t)row * (2 * DIN) + DIN + c];
        ymod[(size_t)row * DIN + c] = y * silu_f(zv);
        __syncthreads();
    }
}

// cls head output: per (b,np): h2(256) @ head_w3^T -> 4, broadcast over P
__global__ __launch_bounds__(256)
void cls_out_kernel(const float* __restrict__ h2, const float* __restrict__ w3,
                    const float* __restrict__ b3, float* __restrict__ out)
{
    __shared__ float red[4 * 256];
    __shared__ float cv[4];
    int blk = blockIdx.x;                 // b*256+np
    int row = blk + (blk >> 8);           // -> b*257+np
    int t = threadIdx.x;
    float hv = h2[(size_t)row * 256 + t];
#pragma unroll
    for (int s = 0; s < 4; ++s) red[s * 256 + t] = hv * w3[s * 256 + t];
    __syncthreads();
    for (int o = 128; o > 0; o >>= 1) {
        if (t < o)
#pragma unroll
            for (int s = 0; s < 4; ++s) red[s * 256 + t] += red[s * 256 + t + o];
        __syncthreads();
    }
    if (t < 4) cv[t] = red[t * 256] + b3[t];
    __syncthreads();
    int p = t & 63, s = t >> 6;
    out[((size_t)blk * PP + p) * SEG + s] = cv[s];
}

// seg final layer: 512 -> 4
__global__ __launch_bounds__(256)
void seg4_kernel(const float* __restrict__ s2, const float* __restrict__ w3,
                 const float* __restrict__ b3, float* __restrict__ out)
{
    int t = threadIdx.x;
    int lr = t >> 2, s = t & 3;
    int row = blockIdx.x * 64 + lr;
    const float* rp = &s2[(size_t)row * 512];
    const float* wp = &w3[(size_t)s * 512];
    float acc = b3[s];
    for (int k = 0; k < 512; ++k) acc += rp[k] * wp[k];
    out[(size_t)row * SEG + s] = acc;
}

// =====================================================================
extern "C" void kernel_launch(void* const* d_in, const int* in_sizes, int n_in,
                              void* d_out, int out_size, void* d_ws, size_t ws_size,
                              hipStream_t stream)
{
    (void)in_sizes; (void)n_in; (void)out_size;
    // ---- inputs ----
    const float* feats    = (const float*)d_in[1];
    const float* centers  = (const float*)d_in[2];
    const float* patch_w  = (const float*)d_in[5];
    const float* patch_b  = (const float*)d_in[6];
    const float* patch_lg = (const float*)d_in[7];
    const float* patch_lb = (const float*)d_in[8];
    const float* face_w1  = (const float*)d_in[9];
    const float* face_b1  = (const float*)d_in[10];
    const float* face_l1g = (const float*)d_in[11];
    const float* face_l1b = (const float*)d_in[12];
    const float* face_w2  = (const float*)d_in[13];
    const float* face_b2  = (const float*)d_in[14];
    const float* face_l2g = (const float*)d_in[15];
    const float* face_l2b = (const float*)d_in[16];
    const float* pos_w1   = (const float*)d_in[17];
    const float* pos_b1   = (const float*)d_in[18];
    const float* pos_w2   = (const float*)d_in[19];
    const float* pos_b2   = (const float*)d_in[20];
    const float* cls_p    = (const float*)d_in[21];
    const float* blk_ln_g = (const float*)d_in[22];
    const float* blk_ln_b = (const float*)d_in[23];
    const float* in_w     = (const float*)d_in[24];
    const float* conv_w   = (const float*)d_in[25];
    const float* conv_b   = (const float*)d_in[26];
    const float* xproj_w  = (const float*)d_in[27];
    const float* dt_w     = (const float*)d_in[28];
    const float* dt_b     = (const float*)d_in[29];
    const float* A_log    = (const float*)d_in[30];
    const float* Dp       = (const float*)d_in[31];
    const float* out_w    = (const float*)d_in[32];
    const float* norm_g   = (const float*)d_in[33];
    const float* norm_b   = (const float*)d_in[34];
    const float* head_w1  = (const float*)d_in[35];
    const float* head_b1  = (const float*)d_in[36];
    const float* head_w2  = (const float*)d_in[37];
    const float* head_b2  = (const float*)d_in[38];
    const float* head_w3  = (const float*)d_in[39];
    const float* head_b3  = (const float*)d_in[40];
    const float* h1_w1    = (const float*)d_in[41];
    const float* h1_b1    = (const float*)d_in[42];
    const float* h1_w2    = (const float*)d_in[43];
    const float* h1_b2    = (const float*)d_in[44];
    const float* h1_w3    = (const float*)d_in[45];
    const float* h1_b3    = (const float*)d_in[46];
    float* out = (float*)d_out;

    // ---- workspace bump allocator (floats) ----
    float* w = (float*)d_ws;
    size_t off = 0;
    auto alloc = [&](size_t n) { float* p = w + off; off += n; return p; };
    float* tokens   = alloc((size_t)ROWS_T * DD);
    float* residual = alloc((size_t)ROWS_T * DD);
    float* hbuf     = alloc((size_t)ROWS_T * DD);
    float* xz       = alloc((size_t)ROWS_T * 2 * DIN);
    float* xc2      = alloc((size_t)ROWS_T * DIN);
    float* xdbl     = alloc((size_t)ROWS_T * 80);
    float* dtb      = alloc((size_t)ROWS_T * DIN);
    float* ymod     = alloc((size_t)ROWS_T * DIN);
    float* posb     = alloc((size_t)ROWS_P * DD);
    float* centerb  = alloc((size_t)ROWS_P * 3 + 4);
    float* patchout = alloc((size_t)ROWS_P * DD);
    float* xfinal   = alloc((size_t)ROWS_T * DD);
    float* hh1      = alloc((size_t)ROWS_T * 512);
    float* hh2      = alloc((size_t)ROWS_T * 256);
    float* outPart  = alloc((size_t)ROWS_T * 1024);
    // chunk buffers for the seg branch, sized from remaining workspace
    size_t remain = (ws_size / 4 > off) ? (ws_size / 4 - off) : 0;
    int CR = 8192;
    const size_t rowF = 768 + 1024 + 1024 + 512;
    while (CR > 64 && (size_t)CR * rowF > remain) CR >>= 1;
    float* cb_ts = alloc((size_t)CR * 768);
    float* cb_g  = alloc((size_t)CR * 1024);
    float* cb_s1 = alloc((size_t)CR * 1024);
    float* cb_s2 = alloc((size_t)CR * 512);

    auto g0 = [&](const float* A, int lda, const float* Wm, int ldw, const float* bias,
                  float* C, int ldc, int M, int N, int Kd) {
        dim3 grid((N + 127) / 128, (M + 127) / 128);
        gemm_kernel<0><<<grid, 256, 0, stream>>>(A, lda, Wm, ldw, bias,
                                                 nullptr, 0, 0, 31, 31, C, ldc, M, N, Kd);
    };
    auto g1 = [&](const float* A, int lda, const float* Wm, int ldw, const float* bias,
                  const float* rowAdd, int ldadd, int rowBase, int shA, int shB,
                  float* C, int ldc, int M, int N, int Kd) {
        dim3 grid((N + 127) / 128, (M + 127) / 128);
        gemm_kernel<1><<<grid, 256, 0, stream>>>(A, lda, Wm, ldw, bias,
                                                 rowAdd, ldadd, rowBase, shA, shB, C, ldc, M, N, Kd);
    };

    // ---- Stage A: token prep ----
    center_kernel<<<(ROWS_P * 3 + 255) / 256, 256, 0, stream>>>(centers, centerb);
    posmlp_kernel<<<ROWS_P, 128, 0, stream>>>(centerb, pos_w1, pos_b1, pos_w2, pos_b2, posb);
    g0(feats, PP * CH, patch_w, PP * CH, patch_b, patchout, DD, ROWS_P, DD, PP * CH);
    // tokens[:, :256] = LN(patchout)+pos  (row b*256+np -> b*257+np via shift 8)
    ln768_kernel<<<ROWS_P, 256, 0, stream>>>(patchout, DD, patch_lg, patch_lb,
                                             posb, DD, tokens, DD, 8);
    set_cls_kernel<<<(BB * DD + 255) / 256, 256, 0, stream>>>(cls_p, tokens);

    // ---- Stage B: 12 mamba blocks ----
    for (int i = 0; i < DEPTH; ++i) {
        residual_ln_kernel<<<ROWS_T, 256, 0, stream>>>(residual, tokens,
                                                       blk_ln_g + (size_t)i * DD,
                                                       blk_ln_b + (size_t)i * DD,
                                                       hbuf, (i == 0) ? 1 : 0);
        g0(hbuf, DD, in_w + (size_t)i * 2 * DIN * DD, DD, nullptr,
           xz, 2 * DIN, ROWS_T, 2 * DIN, DD);
        conv_silu_kernel<<<(ROWS_T * DIN + 255) / 256, 256, 0, stream>>>(
            xz, conv_w + (size_t)i * DIN * KK, conv_b + (size_t)i * DIN, xc2);
        g0(xc2, DIN, xproj_w + (size_t)i * 80 * DIN, DIN, nullptr,
           xdbl, 80, ROWS_T, 80, DIN);
        {   // dt = softplus(xdbl[:, :48] @ dt_w^T + dt_b)
            dim3 grid((DIN + 127) / 128, (ROWS_T + 127) / 128);
            gemm_kernel<4><<<grid, 256, 0, stream>>>(xdbl, 80,
                dt_w + (size_t)i * DIN * RR, RR, dt_b + (size_t)i * DIN,
                nullptr, 0, 0, 31, 31, dtb, DIN, ROWS_T, DIN, RR);
        }
        scan_kernel<<<dim3(DIN / 256, BB), 256, 0, stream>>>(
            dtb, xc2, xdbl, xz, A_log + (size_t)i * DIN * NN, Dp + (size_t)i * DIN, ymod);
        g0(ymod, DIN, out_w + (size_t)i * DD * DIN, DIN, nullptr,
           tokens, DD, ROWS_T, DD, DIN);
    }

    // ---- Stage C: final LN + heads ----
    ln768_kernel<<<ROWS_T, 256, 0, stream>>>(tokens, DD, norm_g, norm_b,
                                             nullptr, 0, xfinal, DD, -1);
    // cls head (per token row; constant over P)
    g1(xfinal, DD, head_w1, DD, head_b1, nullptr, 0, 0, 31, 31, hh1, 512, ROWS_T, 512, DD);
    g1(hh1, 512, head_w2, 512, head_b2, nullptr, 0, 0, 31, 31, hh2, 256, ROWS_T, 256, 512);
    cls_out_kernel<<<ROWS_P, 256, 0, stream>>>(hh2, head_w3, head_b3, out);
    // outcome part of seg head: outPart[b*257+np, :] = x @ h1_w1[:, 768:]^T (no bias here)
    g0(xfinal, DD, h1_w1 + DD, 2 * DD, nullptr, outPart, 1024, ROWS_T, 1024, DD);

    // ---- Stage D: seg branch, chunked over the 131072 face rows ----
    float* out_seg = out + (size_t)ROWS_S * SEG;
    for (long cs = 0; cs < ROWS_S; cs += CR) {
        int rows = (int)(((long)CR < (ROWS_S - cs)) ? CR : (ROWS_S - cs));
        ts_kernel<<<rows, 256, 0, stream>>>(feats + (size_t)cs * CH,
                                            face_w1, face_b1, face_l1g, face_l1b, cb_ts);
        g0(cb_ts, DD, face_w2, DD, face_b2, cb_g, DD, rows, DD, DD);
        ln768_kernel<<<rows, 256, 0, stream>>>(cb_g, DD, face_l2g, face_l2b,
                                               nullptr, 0, cb_ts, DD, -1);
        // s1 = relu( tseg @ h1_w1[:, :768]^T + h1_b1 + outPart[b*257+np] )
        g1(cb_ts, DD, h1_w1, 2 * DD, h1_b1, outPart, 1024, (int)cs, 6, 14,
           cb_s1, 1024, rows, 1024, DD);
        g1(cb_s1, 1024, h1_w2, 1024, h1_b2, nullptr, 0, 0, 31, 31,
           cb_s2, 512, rows, 512, 1024);
        seg4_kernel<<<rows / 64, 256, 0, stream>>>(cb_s2, h1_w3, h1_b3,
                                                   out_seg + (size_t)cs * SEG);
    }
}
